// SingleHeadAttention_79001628443064
// MI455X (gfx1250) — matrix operand order repaired and checked by reading, not compiled, not run on hardware
//
#include <hip/hip_runtime.h>

typedef __attribute__((ext_vector_type(16))) _Float16 v16h;
typedef __attribute__((ext_vector_type(8)))  float    v8f;

#define B_ 4
#define S_ 4096
#define D_ 1024
#define A_ 128

static __device__ __forceinline__ v8f wmma_f16(v16h a, v16h b, v8f c) {
  // D = A(16x32 f16) * B(32x16 f16) + C(16x16 f32)
  return __builtin_amdgcn_wmma_f32_16x16x32_f16(
      /*neg_a=*/false, a, /*neg_b=*/false, b,
      /*c_mod=*/(short)0, c, /*reuse_a=*/false, /*reuse_b=*/false);
}

// Async f16 tile copy: 16 bytes per lane, global -> LDS, tracked by ASYNCcnt.
static __device__ __forceinline__ void async_copy_b128(
    unsigned lds_byte_addr, unsigned global_byte_off, const _Float16* sbase) {
  asm volatile("global_load_async_to_lds_b128 %0, %1, %2"
               :: "v"(lds_byte_addr), "v"(global_byte_off), "s"(sbase)
               : "memory");
}

// ---------------------------------------------------------------------------
// Kernel 1: QKV projection.  [16384,1024] x [1024,128] -> f16 [16384,128]
// grid = (128 row-tiles, 3 weights), block = 256 (8 waves)
// ---------------------------------------------------------------------------
__global__ __launch_bounds__(256) void qkv_proj_kernel(
    const float* __restrict__ E,  const float* __restrict__ Wq,
    const float* __restrict__ Wk, const float* __restrict__ Wv,
    _Float16* __restrict__ Qh, _Float16* __restrict__ Kh, _Float16* __restrict__ Vh)
{
  __shared__ _Float16 Elds[128][40];  // 128 rows x 32 k (+pad)
  __shared__ _Float16 Wt[128][40];    // transposed weight tile: [col][k] (+pad)

  const int tid  = threadIdx.x;
  const int lane = tid & 31;
  const int wave = tid >> 5;
  const int half = lane >> 4;
  const int ml   = lane & 15;

  const float* W  = (blockIdx.y == 0) ? Wq : (blockIdx.y == 1) ? Wk : Wv;
  _Float16*   Out = (blockIdx.y == 0) ? Qh : (blockIdx.y == 1) ? Kh : Vh;

  const int row0 = blockIdx.x * 128;

  v8f acc[8];
  #pragma unroll
  for (int t = 0; t < 8; ++t)
    #pragma unroll
    for (int j = 0; j < 8; ++j) acc[t][j] = 0.0f;

  for (int kt = 0; kt < D_ / 32; ++kt) {
    // stage E tile 128x32 (f32 -> f16), coalesced
    for (int idx = tid; idx < 128 * 32; idx += 256) {
      int r = idx >> 5, c = idx & 31;
      Elds[r][c] = (_Float16)E[(size_t)(row0 + r) * D_ + kt * 32 + c];
    }
    // stage W tile 32x128 transposed -> Wt[col][k]
    for (int idx = tid; idx < 32 * 128; idx += 256) {
      int r = idx >> 7, c = idx & 127;
      Wt[c][r] = (_Float16)W[(size_t)(kt * 32 + r) * A_ + c];
    }
    __syncthreads();

    // A-fragment: rows = lane&15 (wave-relative), K pattern per ISA 7.12.2
    v16h a;
    #pragma unroll
    for (int i = 0; i < 16; ++i)
      a[i] = Elds[wave * 16 + ml][(i & 7) + ((i >> 3) << 4) + (half << 3)];

    // batch-preload all 8 B fragments, then issue 8 independent WMMAs
    v16h bf[8];
    #pragma unroll
    for (int nt = 0; nt < 8; ++nt)
      #pragma unroll
      for (int i = 0; i < 16; ++i)
        bf[nt][i] = Wt[nt * 16 + ml][(i & 7) + ((i >> 3) << 4) + (half << 3)];
    #pragma unroll
    for (int nt = 0; nt < 8; ++nt)
      acc[nt] = wmma_f16(a, bf[nt], acc[nt]);

    __syncthreads();
  }

  #pragma unroll
  for (int nt = 0; nt < 8; ++nt)
    #pragma unroll
    for (int r = 0; r < 8; ++r) {
      int row = row0 + wave * 16 + r + half * 8;   // C/D layout
      int col = nt * 16 + ml;
      Out[(size_t)row * A_ + col] = (_Float16)acc[nt][r];
    }
}

// ---------------------------------------------------------------------------
// Kernel 2: causal flash attention over f16 Q/K/V, f32 output.
// grid = (S/64 query tiles, B), block = 128 (4 waves, 16 queries each)
// ---------------------------------------------------------------------------
__global__ __launch_bounds__(128) void flash_attn_kernel(
    const _Float16* __restrict__ Qh, const _Float16* __restrict__ Kh,
    const _Float16* __restrict__ Vh, float* __restrict__ Out)
{
  __shared__ _Float16 Qlds[64][136];     // 64 queries x 128 d (+pad)
  __shared__ _Float16 Klds[64][136];     // 64 keys x 128 d (+pad)
  __shared__ _Float16 Vt[128][72];       // transposed V tile: [col][key] (+pad)
  __shared__ _Float16 Plds[4][16][72];   // per-wave P scratch: 16 q x 64 keys

  const int tid  = threadIdx.x;
  const int lane = tid & 31;
  const int wave = tid >> 5;
  const int half = lane >> 4;
  const int ml   = lane & 15;

  const int b     = blockIdx.y;
  const int qt    = blockIdx.x;
  const int qrow0 = qt * 64;
  const size_t base = (size_t)b * S_ * A_;

  // stage Q tile once via async copy (f16 -> f16 straight copy)
  {
    const _Float16* qsrc = Qh + base;
    for (int idx = tid; idx < 64 * 16; idx += 128) {
      int r = idx >> 4, cg = (idx & 15) << 3;           // 8 halfs = 16 bytes
      unsigned lds = (unsigned)(uintptr_t)&Qlds[r][cg];
      unsigned go  = (unsigned)(((qrow0 + r) * A_ + cg) * 2);
      async_copy_b128(lds, go, qsrc);
    }
    asm volatile("s_wait_asynccnt 0x0" ::: "memory");
  }
  __syncthreads();

  // Q A-fragments for the 4 d-chunks of 32, held in registers all loop long
  v16h qf[4];
  #pragma unroll
  for (int ch = 0; ch < 4; ++ch)
    #pragma unroll
    for (int i = 0; i < 16; ++i)
      qf[ch][i] = Qlds[wave * 16 + ml][ch * 32 + (i & 7) + ((i >> 3) << 4) + (half << 3)];

  float mrun[8], lrun[8];
  v8f o[8];
  #pragma unroll
  for (int r = 0; r < 8; ++r) { mrun[r] = -__builtin_inff(); lrun[r] = 0.0f; }
  #pragma unroll
  for (int t = 0; t < 8; ++t)
    #pragma unroll
    for (int j = 0; j < 8; ++j) o[t][j] = 0.0f;

  const float scl = 0.08838834764831845f;  // 1/sqrt(128)

  for (int jt = 0; jt <= qt; ++jt) {       // causal: skip upper triangle
    __syncthreads();
    // K tile: async straight copy (row major)
    {
      const _Float16* ksrc = Kh + base;
      for (int idx = tid; idx < 64 * 16; idx += 128) {
        int r = idx >> 4, cg = (idx & 15) << 3;
        unsigned lds = (unsigned)(uintptr_t)&Klds[r][cg];
        unsigned go  = (unsigned)(((jt * 64 + r) * A_ + cg) * 2);
        async_copy_b128(lds, go, ksrc);
      }
    }
    // V tile: element-wise transpose, regular path (overlaps with async)
    for (int idx = tid; idx < 64 * 128; idx += 128) {
      int r = idx >> 7, c = idx & 127;
      Vt[c][r] = Vh[base + (size_t)(jt * 64 + r) * A_ + c];
    }
    asm volatile("s_wait_asynccnt 0x0" ::: "memory");
    __syncthreads();

    // S = Q K^T : d-chunk outer so the 4 inner WMMAs hit independent accums
    v8f s[4];
    #pragma unroll
    for (int st = 0; st < 4; ++st)
      #pragma unroll
      for (int j = 0; j < 8; ++j) s[st][j] = 0.0f;
    #pragma unroll
    for (int ch = 0; ch < 4; ++ch) {
      v16h kb[4];
      #pragma unroll
      for (int st = 0; st < 4; ++st)
        #pragma unroll
        for (int i = 0; i < 16; ++i)
          kb[st][i] = Klds[st * 16 + ml][ch * 32 + (i & 7) + ((i >> 3) << 4) + (half << 3)];
      #pragma unroll
      for (int st = 0; st < 4; ++st)
        s[st] = wmma_f16(qf[ch], kb[st], s[st]);
    }

    // scale + causal mask (branchless select, EXEC stays all-ones)
    const int q_lane_row = qrow0 + wave * 16 + half * 8;  // + r
    const bool diag = (jt == qt);
    #pragma unroll
    for (int st = 0; st < 4; ++st) {
      int key = jt * 64 + st * 16 + ml;
      #pragma unroll
      for (int r = 0; r < 8; ++r) {
        float v = s[st][r] * scl;
        s[st][r] = (diag && key > q_lane_row + r) ? -__builtin_inff() : v;
      }
    }

    // row max over 64 keys: half-wave (16 lane) xor reduction matches C layout
    float alpha[8];
    #pragma unroll
    for (int r = 0; r < 8; ++r) {
      float m0 = fmaxf(fmaxf(s[0][r], s[1][r]), fmaxf(s[2][r], s[3][r]));
      #pragma unroll
      for (int off = 8; off >= 1; off >>= 1)
        m0 = fmaxf(m0, __shfl_xor(m0, off, 16));
      float mnew = fmaxf(mrun[r], m0);
      alpha[r] = __expf(mrun[r] - mnew);   // exp(-inf) = 0 on first tile
      mrun[r] = mnew;
    }

    // P = exp(S - m), row sums, spill P (f16) to per-wave LDS scratch
    float rs[8];
    #pragma unroll
    for (int r = 0; r < 8; ++r) rs[r] = 0.0f;
    #pragma unroll
    for (int st = 0; st < 4; ++st)
      #pragma unroll
      for (int r = 0; r < 8; ++r) {
        float p = __expf(s[st][r] - mrun[r]);
        rs[r] += p;
        Plds[wave][r + half * 8][st * 16 + ml] = (_Float16)p;
      }
    #pragma unroll
    for (int r = 0; r < 8; ++r) {
      #pragma unroll
      for (int off = 8; off >= 1; off >>= 1)
        rs[r] += __shfl_xor(rs[r], off, 16);
      lrun[r] = lrun[r] * alpha[r] + rs[r];
    }

    // rescale running O
    #pragma unroll
    for (int t = 0; t < 8; ++t)
      #pragma unroll
      for (int r = 0; r < 8; ++r) o[t][r] *= alpha[r];

    // same-wave LDS RAW fence before re-reading P as A-fragments
    asm volatile("s_wait_dscnt 0" ::: "memory");

    // O += P V : 2 key-chunks of 32; groups of 4 preloaded B frags ->
    // 4 independent WMMAs per group
    #pragma unroll
    for (int ch = 0; ch < 2; ++ch) {
      v16h pa;
      #pragma unroll
      for (int i = 0; i < 16; ++i)
        pa[i] = Plds[wave][ml][ch * 32 + (i & 7) + ((i >> 3) << 4) + (half << 3)];
      #pragma unroll
      for (int g = 0; g < 2; ++g) {
        v16h vbf[4];
        #pragma unroll
        for (int t = 0; t < 4; ++t)
          #pragma unroll
          for (int i = 0; i < 16; ++i)
            vbf[t][i] = Vt[(g * 4 + t) * 16 + ml][ch * 32 + (i & 7) + ((i >> 3) << 4) + (half << 3)];
        #pragma unroll
        for (int t = 0; t < 4; ++t)
          o[g * 4 + t] = wmma_f16(pa, vbf[t], o[g * 4 + t]);
      }
    }
  }

  // epilogue: normalize, round to 4 decimals (round-to-nearest-even), store
  #pragma unroll
  for (int ct = 0; ct < 8; ++ct)
    #pragma unroll
    for (int r = 0; r < 8; ++r) {
      int row = qrow0 + wave * 16 + r + half * 8;
      float v = o[ct][r] / lrun[r];
      v = rintf(v * 1.0e4f) * 1.0e-4f;
      Out[base + (size_t)row * A_ + ct * 16 + ml] = v;
    }
}

// ---------------------------------------------------------------------------
extern "C" void kernel_launch(void* const* d_in, const int* in_sizes, int n_in,
                              void* d_out, int out_size, void* d_ws, size_t ws_size,
                              hipStream_t stream) {
  const float* E  = (const float*)d_in[0];
  const float* Wq = (const float*)d_in[1];
  const float* Wk = (const float*)d_in[2];
  const float* Wv = (const float*)d_in[3];
  float* out = (float*)d_out;

  _Float16* ws = (_Float16*)d_ws;
  const size_t n = (size_t)B_ * S_ * A_;  // 2,097,152 elements each
  _Float16* Qh = ws;
  _Float16* Kh = ws + n;
  _Float16* Vh = ws + 2 * n;

  qkv_proj_kernel<<<dim3((B_ * S_) / 128, 3), 256, 0, stream>>>(
      E, Wq, Wk, Wv, Qh, Kh, Vh);
  flash_attn_kernel<<<dim3(S_ / 64, B_), 128, 0, stream>>>(Qh, Kh, Vh, out);
}
